// LSTM_generator_26362509263241
// MI455X (gfx1250) — matrix operand order repaired and checked by reading, not compiled, not run on hardware
//
#include <hip/hip_runtime.h>
#include <hip/hip_bf16.h>

typedef __attribute__((ext_vector_type(16))) __bf16 v16bf;
typedef __attribute__((ext_vector_type(8)))  __bf16 v8bf;
typedef __attribute__((ext_vector_type(8)))  float   v8f;

#define HIDDEN 1024
#define GATES  4096
#define BATCH  128
#define TSTEPS 256
#define PIECE  128
#define SONG   32768

// padded LDS row stride (elements). 40 elems = 80 B: 16B-aligned chunks and
// lanes 0..15 hit 16 distinct banks ((n*20) mod 64 all distinct for n<16).
#define BSTRIDE 40

// ---------------- helpers ----------------

__device__ __forceinline__ unsigned short f32_to_bf16(float f) {
    unsigned int u = __float_as_uint(f);
    u += 0x7FFFu + ((u >> 16) & 1u);   // round-to-nearest-even
    return (unsigned short)(u >> 16);
}

__device__ __forceinline__ float sigmoid_f(float x) {
    return 1.0f / (1.0f + __expf(-x));
}

__device__ __forceinline__ float tanh_f(float x) {
    x = fminf(15.0f, fmaxf(-15.0f, x));
    float e = __expf(2.0f * x);
    return (e - 1.0f) / (e + 1.0f);
}

__device__ __forceinline__ v16bf combine16(v8bf lo, v8bf hi) {
    v16bf r;
#pragma unroll
    for (int i = 0; i < 8; ++i) { r[i] = lo[i]; r[i + 8] = hi[i]; }
    return r;
}

// A tile 16x32 (M x K), bf16, row-major with stride ld.
// Layout: lane m (0-15): K=0..15; lane m+16: K=16..31 (matches B's contiguous order)
// -> two contiguous 8-element (16B) chunks per lane.
__device__ __forceinline__ v16bf load_a_tile(const unsigned short* base, int ld,
                                             int mbase, int kbase, int lane) {
    int m    = mbase + (lane & 15);
    int koff = (lane >> 4) << 4;  // 0 or 16
    const v8bf* p0 = reinterpret_cast<const v8bf*>(base + m * ld + kbase + koff);
    const v8bf* p1 = reinterpret_cast<const v8bf*>(base + m * ld + kbase + koff + 8);
    return combine16(*p0, *p1);
}

// B tile 32x16 (K x N). Column n of B == row (jbase+n) of weight matrix [rows x ld]
// (computes X @ W^T). Lanes 0-15 hold K=0..15 of col n, lanes 16-31 hold K=16..31.
__device__ __forceinline__ v16bf load_b_tile(const unsigned short* w, int ld,
                                             int jbase, int kbase, int lane) {
    int n = lane & 15;
    int k = kbase + ((lane >> 4) << 4);  // 0 or 16
    const v8bf* p = reinterpret_cast<const v8bf*>(w + (jbase + n) * ld + k);
    return combine16(p[0], p[1]);
}

// B fragment from an LDS-staged 64-row x 32-K chunk (padded stride BSTRIDE).
__device__ __forceinline__ v16bf load_b_lds(const unsigned short* stage,
                                            int gate, int lane) {
    int n = lane & 15;
    int k = (lane >> 4) << 4;  // 0 or 16
    const v8bf* p = reinterpret_cast<const v8bf*>(stage + (gate * 16 + n) * BSTRIDE + k);
    return combine16(p[0], p[1]);
}

// ---------------- prep kernels ----------------

__global__ __launch_bounds__(256) void cvt_bf16_kernel(const float* __restrict__ src,
                                                       unsigned short* __restrict__ dst,
                                                       int n) {
    int i = blockIdx.x * blockDim.x + threadIdx.x;
    if (i < n) dst[i] = f32_to_bf16(src[i]);
}

__global__ __launch_bounds__(256) void gate_bias_kernel(const float* __restrict__ b_ih,
                                                        const float* __restrict__ b_hh,
                                                        float* __restrict__ gbias) {
    int j = blockIdx.x * blockDim.x + threadIdx.x;
    if (j < GATES) gbias[j] = b_ih[j] + b_hh[j];
}

// step-0 full bias: z @ w_ih^T + b_ih + b_hh   ([BATCH x GATES])
__global__ __launch_bounds__(256) void gate_bias0_kernel(const float* __restrict__ z,
                                                         const float* __restrict__ w_ih,
                                                         const float* __restrict__ b_ih,
                                                         const float* __restrict__ b_hh,
                                                         float* __restrict__ gbias0) {
    int id = blockIdx.x * blockDim.x + threadIdx.x;  // BATCH*GATES threads
    if (id >= BATCH * GATES) return;
    int b = id / GATES;
    int j = id % GATES;
    const float* zr = z + b * 128;          // z is [B,1,128]
    const float* wr = w_ih + j * 128;       // w_ih is [4096,128]
    float acc = b_ih[j] + b_hh[j];
#pragma unroll 8
    for (int k = 0; k < 128; ++k) acc += zr[k] * wr[k];
    gbias0[id] = acc;
}

// ---------------- fused LSTM cell step ----------------
// grid: 64 blocks x 256 threads (8 wave32s).
// block -> ctile = blockIdx.x (16 hidden columns, same for all 8 waves)
// wave  -> mtile = wave       (16 batch rows)
// All 8 waves share the same B tiles (4 gates x 16 rows of w_hh), so the block
// stages each 64x32 B chunk in LDS (double-buffered, one barrier per chunk),
// cutting L2 weight traffic 8x. Each wave accumulates its 16x16 slice of all
// four gates and applies the cell nonlinearity fully in-register.
__global__ __launch_bounds__(256) void lstm_step_kernel(
    const unsigned short* __restrict__ whh,    // [4096][1024] bf16
    const unsigned short* __restrict__ h_in,   // [128][1024] bf16 (unused at t==0)
    unsigned short* __restrict__ h_out,        // [128][1024] bf16
    float* __restrict__ cstate,                // [128][1024] fp32 (in-place)
    const float* __restrict__ gbias,           // [4096]        (t > 0)
    const float* __restrict__ gbias0,          // [128][4096]   (t == 0)
    int t) {
    __shared__ unsigned short bstage[2][64 * BSTRIDE];

    const int tid   = threadIdx.x;
    const int lane  = tid & 31;
    const int wave  = tid >> 5;
    const int cbase = blockIdx.x * 16;   // hidden-column tile (shared by block)
    const int mbase = wave * 16;         // batch-row tile (per wave)
    const int n     = lane & 15;
    const int moff  = (lane >> 4) << 3;  // 0 or 8 (C/D layout)

    v8f acc[4];  // i, f, g, o for (m = mbase+v+moff, idx = cbase+n)

    if (t == 0) {
        // h_prev == 0: gates are exactly the precomputed step-0 bias.
#pragma unroll
        for (int g = 0; g < 4; ++g) {
            int j = g * HIDDEN + cbase + n;
#pragma unroll
            for (int v = 0; v < 8; ++v)
                acc[g][v] = gbias0[(mbase + v + moff) * GATES + j];
        }
    } else {
#pragma unroll
        for (int g = 0; g < 4; ++g) {
            float gb = gbias[g * HIDDEN + cbase + n];
#pragma unroll
            for (int v = 0; v < 8; ++v) acc[g][v] = gb;
        }

        // cooperative B staging: 256 threads x 16B = 4 KB = one 64x32 chunk.
        const int srow  = tid >> 2;          // 0..63  (gate*16 + r)
        const int sk    = (tid & 3) << 3;    // 0,8,16,24
        const int sgate = srow >> 4;
        const int sr    = srow & 15;
        const unsigned short* gsrc =
            whh + (sgate * HIDDEN + cbase + sr) * HIDDEN + sk;  // + kb per chunk
        unsigned short* sdst0 = &bstage[0][srow * BSTRIDE + sk];
        unsigned short* sdst1 = &bstage[1][srow * BSTRIDE + sk];

        // prologue: stage chunk 0 into buffer 0
        *reinterpret_cast<v8bf*>(sdst0) = *reinterpret_cast<const v8bf*>(gsrc);
        __syncthreads();

        for (int kb = 0; kb < HIDDEN; kb += 32) {
            const int buf = (kb >> 5) & 1;
            // issue next chunk's global load early (latency hides under WMMAs)
            v8bf nxt = {};
            const bool have_next = (kb + 32) < HIDDEN;
            if (have_next)
                nxt = *reinterpret_cast<const v8bf*>(gsrc + kb + 32);

            v16bf a = load_a_tile(h_in, HIDDEN, mbase, kb, lane);
            const unsigned short* stage = &bstage[buf][0];
            // hoist all four B fragments into distinct registers so the
            // scheduler can batch the 8 ds_load_b128 and stagger dscnt waits,
            // letting the 4 independent-accumulator WMMAs issue back-to-back.
            v16bf b0 = load_b_lds(stage, 0, lane);
            v16bf b1 = load_b_lds(stage, 1, lane);
            v16bf b2 = load_b_lds(stage, 2, lane);
            v16bf b3 = load_b_lds(stage, 3, lane);
            acc[0] = __builtin_amdgcn_wmma_f32_16x16x32_bf16(
                false, a, false, b0, (short)0, acc[0], false, false);
            acc[1] = __builtin_amdgcn_wmma_f32_16x16x32_bf16(
                false, a, false, b1, (short)0, acc[1], false, false);
            acc[2] = __builtin_amdgcn_wmma_f32_16x16x32_bf16(
                false, a, false, b2, (short)0, acc[2], false, false);
            acc[3] = __builtin_amdgcn_wmma_f32_16x16x32_bf16(
                false, a, false, b3, (short)0, acc[3], false, false);

            // write next chunk into the other buffer; one barrier keeps the
            // double-buffer invariant (reads of buf / writes of buf^1 disjoint).
            if (have_next)
                *reinterpret_cast<v8bf*>(buf ? sdst0 : sdst1) = nxt;
            __syncthreads();
        }
    }

    // elementwise cell update (each (m, idx) owned by exactly one lane/vgpr)
    const int idx = cbase + n;
#pragma unroll
    for (int v = 0; v < 8; ++v) {
        int m = mbase + v + moff;
        float ig = sigmoid_f(acc[0][v]);
        float fg = sigmoid_f(acc[1][v]);
        float gg = tanh_f(acc[2][v]);
        float og = sigmoid_f(acc[3][v]);
        float c_old = (t == 0) ? 0.0f : cstate[m * HIDDEN + idx];
        float c_new = fg * c_old + ig * gg;
        float h_new = og * tanh_f(c_new);
        cstate[m * HIDDEN + idx] = c_new;
        h_out[m * HIDDEN + idx]  = f32_to_bf16(h_new);
    }
}

// ---------------- FC projection: y_t = h_t @ W_fc^T + b_fc ----------------
// grid: 8 blocks x 256 threads = 64 waves; one 16x16 output tile per wave.
__global__ __launch_bounds__(256) void fc_step_kernel(
    const unsigned short* __restrict__ wfc,   // [128][1024] bf16
    const unsigned short* __restrict__ h,     // [128][1024] bf16
    const float* __restrict__ b_fc,           // [128]
    float* __restrict__ out,                  // [128][32768]
    int t) {
    const int lane  = threadIdx.x & 31;
    const int wave  = threadIdx.x >> 5;
    const int wid   = blockIdx.x * 8 + wave;  // 0..63
    const int mbase = (wid >> 3) * 16;
    const int nbase = (wid & 7) * 16;
    const int n     = lane & 15;
    const int moff  = (lane >> 4) << 3;

    float bb = b_fc[nbase + n];
    v8f acc;
#pragma unroll
    for (int v = 0; v < 8; ++v) acc[v] = bb;

    for (int kb = 0; kb < HIDDEN; kb += 32) {
        v16bf a = load_a_tile(h, HIDDEN, mbase, kb, lane);
        v16bf b = load_b_tile(wfc, HIDDEN, nbase, kb, lane);
        acc = __builtin_amdgcn_wmma_f32_16x16x32_bf16(
            false, a, false, b, (short)0, acc, false, false);
    }

#pragma unroll
    for (int v = 0; v < 8; ++v) {
        int m = mbase + v + moff;
        out[m * SONG + t * PIECE + nbase + n] = acc[v];
    }
}

// ---------------- host side ----------------

extern "C" void kernel_launch(void* const* d_in, const int* in_sizes, int n_in,
                              void* d_out, int out_size, void* d_ws, size_t ws_size,
                              hipStream_t stream) {
    (void)in_sizes; (void)n_in; (void)out_size; (void)ws_size;

    // input order: current_batch_size, z, w_ih, w_hh, b_ih, b_hh, W_fc, b_fc
    const float* z    = (const float*)d_in[1];   // [128,1,128]
    const float* w_ih = (const float*)d_in[2];   // [4096,128]
    const float* w_hh = (const float*)d_in[3];   // [4096,1024]
    const float* b_ih = (const float*)d_in[4];   // [4096]
    const float* b_hh = (const float*)d_in[5];   // [4096]
    const float* W_fc = (const float*)d_in[6];   // [128,1024]
    const float* b_fc = (const float*)d_in[7];   // [128]
    float* out = (float*)d_out;                  // [128,32768]

    // scratch layout (256B aligned slabs)
    char* ws = (char*)d_ws;
    unsigned short* whh_bf = (unsigned short*)(ws + 0);          //  8,388,608 B
    unsigned short* wfc_bf = (unsigned short*)(ws + 8388608);    //    262,144 B
    float*          gbias0 = (float*)(ws + 8650752);             //  2,097,152 B
    float*          gbias  = (float*)(ws + 10747904);            //     16,384 B
    unsigned short* hbuf   = (unsigned short*)(ws + 10764288);   //    524,288 B (x2)
    float*          cbuf   = (float*)(ws + 11288576);            //    524,288 B

    // prep (deterministic, rerun every call)
    cvt_bf16_kernel<<<(GATES * HIDDEN + 255) / 256, 256, 0, stream>>>(w_hh, whh_bf,
                                                                      GATES * HIDDEN);
    cvt_bf16_kernel<<<(PIECE * HIDDEN + 255) / 256, 256, 0, stream>>>(W_fc, wfc_bf,
                                                                      PIECE * HIDDEN);
    gate_bias_kernel<<<(GATES + 255) / 256, 256, 0, stream>>>(b_ih, b_hh, gbias);
    gate_bias0_kernel<<<(BATCH * GATES + 255) / 256, 256, 0, stream>>>(z, w_ih, b_ih,
                                                                       b_hh, gbias0);

    // sequential recurrence: 2 kernels per step, double-buffered h
    for (int t = 0; t < TSTEPS; ++t) {
        unsigned short* h_in  = hbuf + ((t + 1) & 1) * (BATCH * HIDDEN);
        unsigned short* h_out = hbuf + (t & 1) * (BATCH * HIDDEN);
        lstm_step_kernel<<<64, 256, 0, stream>>>(whh_bf, h_in, h_out, cbuf,
                                                 gbias, gbias0, t);
        fc_step_kernel<<<8, 256, 0, stream>>>(wfc_bf, h_out, b_fc, out, t);
    }
}
